// BaseModel_3624952398338
// MI455X (gfx1250) — matrix operand order repaired and checked by reading, not compiled, not run on hardware
//
#include <hip/hip_runtime.h>
#include <hip/hip_bf16.h>
#include <math.h>

// ---------------- WMMA types (CDNA5 / gfx1250, wave32) ----------------
typedef __attribute__((ext_vector_type(16))) _Float16 v16h;
typedef __attribute__((ext_vector_type(8)))  _Float16 v8h;
typedef __attribute__((ext_vector_type(8)))  float    v8f;
static_assert(sizeof(v16h) == 32, "v16h");

#define NNODES 40000
#define NEDGES 600000
#define NGRAPH 32
#define HP 48   // padded stride for vh scratch (max H = 33)

// Per-lane async gather: global f32 -> LDS f32 (tracked by ASYNCcnt).
__device__ __forceinline__ void async_gather_b32(float* ldsDst, const float* gSrc)
{
  unsigned lo = (unsigned)(size_t)ldsDst;
  unsigned long long ga = (unsigned long long)(size_t)gSrc;
  asm volatile("global_load_async_to_lds_b32 %0, %1, off"
               :: "v"(lo), "v"(ga) : "memory");
}
__device__ __forceinline__ void wait_async()
{
  asm volatile("s_wait_asynccnt 0" ::: "memory");
}

// =====================================================================
// Generic GVP stage: one 128-thread block, MT 16-row tiles (R = MT*16 rows).
//   s_out = A(sIn cat vn) @ Ws + b   (WMMA f16->f32, B reused across MT)
//   gate  = sigmoid(s_out @ Wsv + b) (WMMA)
//   v_out = (vh @ Wv) * gate         (VALU)
// OUTMODE: 0 = chain in LDS, 1 = atomicAdd scatter (edges), 2 = store (nodes)
// =====================================================================
template<int MT,int SI,int VI,int H,int SO,int VO,int LDI,int LDO,bool RELU,int OUTMODE>
__device__ __forceinline__ void gvp_stage(
    _Float16* __restrict__ sIn, const float* __restrict__ vIn,
    _Float16* __restrict__ sOut, float* __restrict__ vOut,
    float* __restrict__ vhB, float* __restrict__ gateB,
    const float* __restrict__ whW, const _Float16* __restrict__ wsP,
    const float* __restrict__ wsB,
    const _Float16* __restrict__ wsvP, const float* __restrict__ wsvB,
    const float* __restrict__ wvW,
    const int* __restrict__ rowIdx, float* __restrict__ gS, float* __restrict__ gV)
{
  const int tid  = threadIdx.x;
  const int lane = tid & 31;
  const int wave = tid >> 5;
  const int hi8  = (lane & 16) ? 8 : 0;
  constexpr int R  = MT * 16;
  constexpr int KT = LDI / 32;            // K tiles for Ws
  constexpr int NT = (SO + 15) / 16;      // N tiles for Ws
  constexpr int GK = (SO + 31) / 32;      // K tiles for gate matmul

  // ---- phase 1: vh = v^T @ Wh  (VALU), and zero sOut pad cols [SO, GK*32)
  for (int t = tid; t < R * 3 * H; t += 128) {
    int m = t / (3 * H), r = t % (3 * H), d = r / H, h = r % H;
    float a = 0.f;
    for (int vi = 0; vi < VI; ++vi)
      a += vIn[(m * VI + vi) * 3 + d] * whW[vi * H + h];
    vhB[(m * 3 + d) * HP + h] = a;
  }
  constexpr int PADC = GK * 32 - SO;
  for (int t = tid; t < R * PADC; t += 128) {
    int m = t / PADC, c = t % PADC;
    sOut[m * LDO + SO + c] = (_Float16)0.f;
  }
  __syncthreads();

  // ---- phase 2: vn -> appended f16 columns of sIn
  for (int t = tid; t < R * H; t += 128) {
    int m = t / H, h = t % H;
    float s2 = 0.f;
    for (int d = 0; d < 3; ++d) { float x = vhB[(m * 3 + d) * HP + h]; s2 += x * x; }
    sIn[m * LDI + SI + h] = (_Float16)sqrtf(fmaxf(s2, 1e-8f));
  }
  __syncthreads();

  // ---- phase 3: s = A @ Ws + b (WMMA, B fragment reused across MT M-tiles)
  for (int nt = wave; nt < NT; nt += 4) {
    v8f acc[MT];
    v8f zacc = {};
#pragma unroll
    for (int mt = 0; mt < MT; ++mt) acc[mt] = zacc;
#pragma unroll
    for (int kt = 0; kt < KT; ++kt) {
      v16h b = *(const v16h*)(wsP + (((size_t)(nt * KT + kt)) << 9) + (lane << 4));
#pragma unroll
      for (int mt = 0; mt < MT; ++mt) {
        const _Float16* ap = sIn + (mt * 16 + (lane & 15)) * LDI + kt * 32 + ((lane & 16) ? 8 : 0);
        v8h alo = *(const v8h*)ap;
        v8h ahi = *(const v8h*)(ap + 16);
        v16h a;
#pragma unroll
        for (int i = 0; i < 8; ++i) { a[i] = alo[i]; a[8 + i] = ahi[i]; }
        acc[mt] = __builtin_amdgcn_wmma_f32_16x16x32_f16(false, a, false, b, (short)0, acc[mt], false, false);
      }
    }
    int n = nt * 16 + (lane & 15);
    if (n < SO) {
#pragma unroll
      for (int mt = 0; mt < MT; ++mt) {
#pragma unroll
        for (int r = 0; r < 8; ++r) {
          int m = mt * 16 + r + hi8;
          float v = acc[mt][r] + wsB[n];
          if (VO == 0 && RELU) v = fmaxf(v, 0.f);
          sOut[m * LDO + n] = (_Float16)v;   // pre-ReLU when VO>0 (gate uses pre-act s)
          if (OUTMODE == 1) {
            int d0 = rowIdx[m];
            if (d0 >= 0) atomicAdd(&gS[(size_t)d0 * SO + n], v);
          } else if (OUTMODE == 2) {
            int d0 = rowIdx[m];
            if (d0 >= 0) gS[(size_t)d0 * SO + n] = v;
          }
        }
      }
    }
  }
  __syncthreads();

  if constexpr (VO > 0) {
    // ---- phase 4: gate = sigmoid(s @ Wsv + b) (WMMA)
    constexpr int NTG = (VO + 15) / 16;
    for (int nt = wave; nt < NTG; nt += 4) {
      v8f acc[MT];
      v8f zacc = {};
#pragma unroll
      for (int mt = 0; mt < MT; ++mt) acc[mt] = zacc;
#pragma unroll
      for (int kt = 0; kt < GK; ++kt) {
        v16h b = *(const v16h*)(wsvP + (((size_t)(nt * GK + kt)) << 9) + (lane << 4));
#pragma unroll
        for (int mt = 0; mt < MT; ++mt) {
          const _Float16* ap = sOut + (mt * 16 + (lane & 15)) * LDO + kt * 32 + ((lane & 16) ? 8 : 0);
          v8h alo = *(const v8h*)ap;
          v8h ahi = *(const v8h*)(ap + 16);
          v16h a;
#pragma unroll
          for (int i = 0; i < 8; ++i) { a[i] = alo[i]; a[8 + i] = ahi[i]; }
          acc[mt] = __builtin_amdgcn_wmma_f32_16x16x32_f16(false, a, false, b, (short)0, acc[mt], false, false);
        }
      }
      int n = nt * 16 + (lane & 15);
      if (n < VO) {
#pragma unroll
        for (int mt = 0; mt < MT; ++mt) {
#pragma unroll
          for (int r = 0; r < 8; ++r) {
            int m = mt * 16 + r + hi8;
            float x = acc[mt][r] + wsvB[n];
            gateB[m * 32 + n] = 1.f / (1.f + expf(-x));
          }
        }
      }
    }
    __syncthreads();
    if (RELU) { // apply ReLU for the chained consumer (gate already used pre-act s)
      for (int t = tid; t < R * SO; t += 128) {
        int m = t / SO, n = t % SO;
        float v = (float)sOut[m * LDO + n];
        sOut[m * LDO + n] = (_Float16)fmaxf(v, 0.f);
      }
      __syncthreads();
    }
    // ---- phase 5: v_out = (vh @ Wv) * gate
    for (int t = tid; t < R * VO * 3; t += 128) {
      int m = t / (VO * 3), r = t % (VO * 3), vo = r / 3, d = r % 3;
      float a = 0.f;
      for (int h = 0; h < H; ++h) a += vhB[(m * 3 + d) * HP + h] * wvW[h * VO + vo];
      float v = a * gateB[m * 32 + vo];
      if (OUTMODE == 1) {
        int d0 = rowIdx[m];
        if (d0 >= 0) atomicAdd(&gV[((size_t)d0 * VO + vo) * 3 + d], v);
      } else if (OUTMODE == 2) {
        int d0 = rowIdx[m];
        if (d0 >= 0) gV[((size_t)d0 * VO + vo) * 3 + d] = v;
      } else {
        vOut[(m * VO + vo) * 3 + d] = v;
      }
    }
    __syncthreads();
  }
}

// =====================================================================
// Weight packing: f32 [K,N] row-major -> f16 WMMA B-fragment tiles.
// Tile t = nt*KT + kt; lane l holds column n = nt*16+(l&15),
// 16 halves k = kt*32 + (l<16?0:8) + (j&7) + (j<8?0:16).
// =====================================================================
__global__ void k_pack(const float* W, _Float16* P, int K, int Nn, int KT, int NT2)
{
  long long idx = (long long)blockIdx.x * blockDim.x + threadIdx.x;
  long long total = (long long)KT * NT2 * 512;
  if (idx >= total) return;
  int w = (int)(idx & 511);
  int t = (int)(idx >> 9);
  int lane = w >> 4, j = w & 15;
  int kt = t % KT, nt = t / KT;
  int n = nt * 16 + (lane & 15);
  int k = kt * 32 + ((lane & 16) ? 8 : 0) + (j & 7) + ((j & 8) ? 16 : 0);
  float v = (k < K && n < Nn) ? W[(size_t)k * Nn + n] : 0.f;
  P[idx] = (_Float16)v;
}

__global__ void k_zero(float* p, long long n)
{
  long long i = (long long)blockIdx.x * blockDim.x + threadIdx.x;
  if (i < n) p[i] = 0.f;
}

__global__ void k_cnt(const int* eidx, float* cnt, int E_)
{
  int e = blockIdx.x * blockDim.x + threadIdx.x;
  if (e < E_) atomicAdd(&cnt[eidx[E_ + e]], 1.f);
}

__global__ void k_gcnt(const int* batch, float* gcnt, int N_)
{
  int n = blockIdx.x * blockDim.x + threadIdx.x;
  if (n < N_) atomicAdd(&gcnt[batch[n]], 1.f);
}

// ---------------- node embedding: LN(embed[atom]) @ Ws + b ----------------
__global__ void k_node_embed(const int* atoms, const float* embed,
                             const float* lng, const float* lnb,
                             const float* wsW, const float* wsB2,
                             float* hs, float* hv, int N_)
{
  int n = blockIdx.x * blockDim.x + threadIdx.x;
  if (n >= N_) return;
  float t[9]; float mu = 0.f;
  int a = atoms[n];
  for (int k = 0; k < 9; ++k) { t[k] = embed[a * 9 + k]; mu += t[k]; }
  mu *= (1.f / 9.f);
  float var = 0.f;
  for (int k = 0; k < 9; ++k) { float d = t[k] - mu; var += d * d; }
  var *= (1.f / 9.f);
  float is = rsqrtf(var + 1e-5f);
  for (int k = 0; k < 9; ++k) t[k] = (t[k] - mu) * is * lng[k] + lnb[k];
  for (int c = 0; c < 100; ++c) {
    float acc = wsB2[c];
    for (int k = 0; k < 9; ++k) acc += t[k] * wsW[k * 100 + c];
    hs[(size_t)n * 100 + c] = acc;
  }
  for (int j = 0; j < 48; ++j) hv[(size_t)n * 48 + j] = 0.f;
}

// ---------------- edge embedding: LN + tiny GVP (si=16,vi=1 -> 32,1) -------
__global__ void k_edge_embed(const float* eS, const float* eV,
                             const float* lng, const float* lnb,
                             const float* wh, const float* wsW, const float* wsB2,
                             const float* wsvW, const float* wsvB2, const float* wvW,
                             float* esOut, float* evOut, int E_)
{
  int e = blockIdx.x * blockDim.x + threadIdx.x;
  if (e >= E_) return;
  float s[16]; float mu = 0.f;
  for (int c = 0; c < 16; ++c) { s[c] = eS[(size_t)e * 16 + c]; mu += s[c]; }
  mu *= (1.f / 16.f);
  float var = 0.f;
  for (int c = 0; c < 16; ++c) { float d = s[c] - mu; var += d * d; }
  var *= (1.f / 16.f);
  float is = rsqrtf(var + 1e-5f);
  for (int c = 0; c < 16; ++c) s[c] = (s[c] - mu) * is * lng[c] + lnb[c];
  float v[3]; float vn2 = 0.f;
  for (int d = 0; d < 3; ++d) { v[d] = eV[(size_t)e * 3 + d]; vn2 += v[d] * v[d]; }
  vn2 = fmaxf(vn2, 1e-8f);
  float ivn = rsqrtf(vn2);
  float vh3[3]; float nh2 = 0.f;
  for (int d = 0; d < 3; ++d) { vh3[d] = v[d] * ivn * wh[0]; nh2 += vh3[d] * vh3[d]; }
  float vnorm = sqrtf(fmaxf(nh2, 1e-8f));
  float ga = 0.f;
  for (int j = 0; j < 32; ++j) {
    float a = wsB2[j] + vnorm * wsW[16 * 32 + j];
    for (int c = 0; c < 16; ++c) a += s[c] * wsW[c * 32 + j];
    esOut[(size_t)e * 32 + j] = a;
    ga += a * wsvW[j];
  }
  float gate = 1.f / (1.f + expf(-(ga + wsvB2[0])));
  for (int d = 0; d < 3; ++d) evOut[(size_t)e * 3 + d] = vh3[d] * wvW[0] * gate;
}

// ---------------- residual + LayerNorm (s 100-dim; v 16x3 norm) ------------
__global__ void k_ln(const float* gw, const float* bw,
                     const float* sIn, const float* vIn,
                     const float* addS, const float* addV, const float* cntp,
                     float* outS, float* outV, int N_)
{
  int n = blockIdx.x * blockDim.x + threadIdx.x;
  if (n >= N_) return;
  float inv = cntp ? 1.f / fmaxf(cntp[n], 1.f) : 1.f;
  size_t sb = (size_t)n * 100, vb = (size_t)n * 48;
  float mu = 0.f;
  for (int c = 0; c < 100; ++c) { float x = sIn[sb + c] + (addS ? addS[sb + c] * inv : 0.f); mu += x; }
  mu *= 0.01f;
  float var = 0.f;
  for (int c = 0; c < 100; ++c) {
    float x = sIn[sb + c] + (addS ? addS[sb + c] * inv : 0.f) - mu; var += x * x;
  }
  var *= 0.01f;
  float is = rsqrtf(var + 1e-5f);
  for (int c = 0; c < 100; ++c) {
    float x = sIn[sb + c] + (addS ? addS[sb + c] * inv : 0.f);
    outS[sb + c] = (x - mu) * is * gw[c] + bw[c];
  }
  float vs = 0.f;
  for (int vo = 0; vo < 16; ++vo) {
    float s2 = 0.f;
    for (int d = 0; d < 3; ++d) {
      float y = vIn[vb + vo * 3 + d] + (addV ? addV[vb + vo * 3 + d] * inv : 0.f);
      s2 += y * y;
    }
    vs += fmaxf(s2, 1e-8f);
  }
  float ivn = rsqrtf(vs * (1.f / 16.f));
  for (int j = 0; j < 48; ++j) {
    float y = vIn[vb + j] + (addV ? addV[vb + j] * inv : 0.f);
    outV[vb + j] = y * ivn;
  }
}

// =====================================================================
// Edge message kernel: 32 edges per block (2 M-tiles), 3 chained GVPs,
// async global->LDS gather for vector features, atomic scatter out.
// =====================================================================
__global__ __launch_bounds__(128) void k_edge_msg(
    const float* __restrict__ hs, const float* __restrict__ hv,
    const float* __restrict__ es, const float* __restrict__ ev,
    const int* __restrict__ eidx, int E_,
    const float* wh0, const _Float16* ws0, const float* wsb0,
    const _Float16* wsv0, const float* wsvb0, const float* wv0,
    const float* wh1, const _Float16* ws1, const float* wsb1,
    const _Float16* wsv1, const float* wsvb1, const float* wv1,
    const float* wh2, const _Float16* ws2, const float* wsb2,
    const _Float16* wsv2, const float* wsvb2, const float* wv2,
    float* aggS, float* aggV)
{
  __shared__ _Float16 sA[32 * 288];
  __shared__ _Float16 sB[32 * 128];
  __shared__ _Float16 sC[32 * 128];
  __shared__ float vhB[32 * 3 * HP];
  __shared__ float v0[32 * 33 * 3];
  __shared__ float v1[32 * 16 * 3];
  __shared__ float v2[32 * 16 * 3];
  __shared__ float gateB[32 * 32];
  __shared__ int rowIdx[32], srcIdx[32];

  const int tid = threadIdx.x;
  const int e0 = blockIdx.x * 32;
  if (tid < 32) {
    int e = e0 + tid;
    rowIdx[tid] = (e < E_) ? eidx[E_ + e] : -1;
    srcIdx[tid] = (e < E_) ? eidx[e] : 0;
  }
  __syncthreads();

  // gather scalar features: [s_src(100) | es(32) | s_dst(100)] + zero pad
  for (int t = tid; t < 32 * 288; t += 128) {
    int m = t / 288, c = t % 288;
    int e = e0 + m;
    float v = 0.f;
    if (e < E_ && c < 232) {
      if (c < 100)       v = hs[(size_t)srcIdx[m] * 100 + c];
      else if (c < 132)  v = es[(size_t)e * 32 + (c - 100)];
      else               v = hs[(size_t)rowIdx[m] * 100 + (c - 132)];
    }
    sA[t] = (_Float16)v;
  }
  // gather vector features [v_src(16) | ev(1) | v_dst(16)] via async->LDS
  for (int t = tid; t < 32 * 33 * 3; t += 128) {
    int m = t / 99, r = t % 99, vi = r / 3, d = r % 3;
    int e = e0 + m;
    const float* gp = nullptr;
    if (e < E_) {
      if (vi < 16)       gp = &hv[((size_t)srcIdx[m] * 16 + vi) * 3 + d];
      else if (vi == 16) gp = &ev[(size_t)e * 3 + d];
      else               gp = &hv[((size_t)rowIdx[m] * 16 + (vi - 17)) * 3 + d];
    }
    if (gp) async_gather_b32(&v0[(m * 33 + vi) * 3 + d], gp);
    else    v0[(m * 33 + vi) * 3 + d] = 0.f;
  }
  for (int t = tid; t < 32 * 128; t += 128) { sB[t] = (_Float16)0.f; sC[t] = (_Float16)0.f; }
  wait_async();
  __syncthreads();

  gvp_stage<2, 232, 33, 33, 100, 16, 288, 128, true, 0>(
      sA, v0, sB, v1, vhB, gateB, wh0, ws0, wsb0, wsv0, wsvb0, wv0, rowIdx, nullptr, nullptr);
  gvp_stage<2, 100, 16, 16, 100, 16, 128, 128, true, 0>(
      sB, v1, sC, v2, vhB, gateB, wh1, ws1, wsb1, wsv1, wsvb1, wv1, rowIdx, nullptr, nullptr);
  gvp_stage<2, 100, 16, 16, 100, 16, 128, 128, false, 1>(
      sC, v2, sB, nullptr, vhB, gateB, wh2, ws2, wsb2, wsv2, wsvb2, wv2, rowIdx, aggS, aggV);
}

// =====================================================================
// Node feed-forward: 32 nodes per block (2 M-tiles), GVP -> GVP -> store
// =====================================================================
__global__ __launch_bounds__(128) void k_ff(
    const float* __restrict__ hs, const float* __restrict__ hv,
    const float* wh0, const _Float16* ws0, const float* wsb0,
    const _Float16* wsv0, const float* wsvb0, const float* wv0,
    const float* wh1, const _Float16* ws1, const float* wsb1,
    const _Float16* wsv1, const float* wsvb1, const float* wv1,
    float* fs, float* fv, int N_)
{
  __shared__ _Float16 sA[32 * 160];
  __shared__ _Float16 sBig[32 * 448];
  __shared__ _Float16 sC[32 * 128];
  __shared__ float vhB[32 * 3 * HP];
  __shared__ float v0[32 * 16 * 3];
  __shared__ float v1[32 * 32 * 3];
  __shared__ float gateB[32 * 32];
  __shared__ int rowIdx[32];

  const int tid = threadIdx.x;
  const int n0 = blockIdx.x * 32;
  if (tid < 32) rowIdx[tid] = (n0 + tid < N_) ? (n0 + tid) : -1;
  for (int t = tid; t < 32 * 160; t += 128) {
    int m = t / 160, c = t % 160;
    int n = n0 + m;
    float v = (n < N_ && c < 100) ? hs[(size_t)n * 100 + c] : 0.f;
    sA[t] = (_Float16)v;
  }
  // hv rows -> LDS via async copy (f32, no conversion needed)
  for (int t = tid; t < 32 * 48; t += 128) {
    int m = t / 48, j = t % 48;
    int n = n0 + m;
    if (n < N_) async_gather_b32(&v0[t], &hv[(size_t)n * 48 + j]);
    else        v0[t] = 0.f;
  }
  for (int t = tid; t < 32 * 448; t += 128) sBig[t] = (_Float16)0.f;
  for (int t = tid; t < 32 * 128; t += 128) sC[t] = (_Float16)0.f;
  wait_async();
  __syncthreads();

  gvp_stage<2, 100, 16, 32, 400, 32, 160, 448, true, 0>(
      sA, v0, sBig, v1, vhB, gateB, wh0, ws0, wsb0, wsv0, wsvb0, wv0, rowIdx, nullptr, nullptr);
  gvp_stage<2, 400, 32, 32, 100, 16, 448, 128, false, 2>(
      sBig, v1, sC, nullptr, vhB, gateB, wh1, ws1, wsb1, wsv1, wsvb1, wv1, rowIdx, fs, fv);
}

// ---------------- output GVP (so=100, vo=0, relu) --------------------------
__global__ __launch_bounds__(128) void k_wout(
    const float* __restrict__ lnS, const float* __restrict__ lnV,
    const float* wh, const _Float16* wsPk, const float* wsb,
    float* outS, int N_)
{
  __shared__ _Float16 sA[16 * 128];
  __shared__ _Float16 sC[16 * 128];
  __shared__ float vhB[16 * 3 * HP];
  __shared__ float v0[16 * 16 * 3];
  __shared__ float gateB[16 * 32];
  __shared__ int rowIdx[16];

  const int tid = threadIdx.x;
  const int n0 = blockIdx.x * 16;
  if (tid < 16) rowIdx[tid] = (n0 + tid < N_) ? (n0 + tid) : -1;
  for (int t = tid; t < 16 * 128; t += 128) {
    int m = t / 128, c = t % 128;
    int n = n0 + m;
    float v = (n < N_ && c < 100) ? lnS[(size_t)n * 100 + c] : 0.f;
    sA[t] = (_Float16)v;
    sC[t] = (_Float16)0.f;
  }
  for (int t = tid; t < 16 * 48; t += 128) {
    int m = t / 48, j = t % 48;
    int n = n0 + m;
    v0[t] = (n < N_) ? lnV[(size_t)n * 48 + j] : 0.f;
  }
  __syncthreads();
  gvp_stage<1, 100, 16, 16, 100, 0, 128, 128, true, 2>(
      sA, v0, sC, nullptr, vhB, gateB, wh, wsPk, wsb,
      nullptr, nullptr, nullptr, rowIdx, outS, nullptr);
}

// ---------------- pooling (batch is sorted -> run-length + atomics) --------
__global__ void k_pool(const float* nodeOut, const int* batch, float* pooled, int N_)
{
  int c = threadIdx.x;
  if (c >= 100) return;
  int n0 = blockIdx.x * 128;
  float acc = 0.f; int cur = -1;
  for (int i = 0; i < 128; ++i) {
    int n = n0 + i;
    if (n >= N_) break;
    int g = batch[n];
    if (g != cur) {
      if (cur >= 0) atomicAdd(&pooled[cur * 100 + c], acc);
      cur = g; acc = 0.f;
    }
    acc += nodeOut[(size_t)n * 100 + c];
  }
  if (cur >= 0) atomicAdd(&pooled[cur * 100 + c], acc);
}

// ---------------- dense head ----------------------------------------------
__global__ void k_dense(const float* pooled, const float* gcnt,
                        const float* w1, const float* b1,
                        const float* w2, const float* b2, float* outp)
{
  __shared__ float pm[32 * 100];
  __shared__ float hb[32 * 200];
  int tid = threadIdx.x;
  for (int t = tid; t < 3200; t += 256) {
    int g = t / 100;
    pm[t] = pooled[t] / fmaxf(gcnt[g], 1.f);
  }
  __syncthreads();
  for (int t = tid; t < 32 * 200; t += 256) {
    int g = t / 200, j = t % 200;
    float a = b1[j];
    for (int k = 0; k < 100; ++k) a += pm[g * 100 + k] * w1[k * 200 + j];
    hb[t] = fmaxf(a, 0.f);
  }
  __syncthreads();
  if (tid < 32) {
    float a = b2[0];
    for (int j = 0; j < 200; ++j) a += hb[tid * 200 + j] * w2[j];
    outp[tid] = a;
  }
}

// =====================================================================
// Host orchestration
// =====================================================================
extern "C" void kernel_launch(void* const* d_in, const int* in_sizes, int n_in,
                              void* d_out, int out_size, void* d_ws, size_t ws_size,
                              hipStream_t stream)
{
  (void)in_sizes; (void)n_in; (void)out_size; (void)ws_size;
  const int N = NNODES, E = NEDGES;
  auto F = [&](int i) { return (const float*)d_in[i]; };
  auto I = [&](int i) { return (const int*)d_in[i]; };
  const int* atoms   = I(0);
  const float* edgeS = F(1);
  const float* edgeV = F(2);
  const int* eidx    = I(3);
  const int* batch   = I(4);

  // ---- workspace bump allocator ----
  char* wp = (char*)d_ws;
  size_t off = 0;
  auto alloc = [&](size_t bytes) -> char* {
    char* p = wp + off;
    off = (off + bytes + 255) & ~(size_t)255;
    return p;
  };
  float* hs      = (float*)alloc((size_t)N * 100 * 4);
  float* hv      = (float*)alloc((size_t)N * 48 * 4);
  float* aggS    = (float*)alloc((size_t)N * 100 * 4);
  float* aggV    = (float*)alloc((size_t)N * 48 * 4);
  float* esb     = (float*)alloc((size_t)E * 32 * 4);
  float* evb     = (float*)alloc((size_t)E * 3 * 4);
  float* cnt     = (float*)alloc((size_t)N * 4);
  float* nodeOut = (float*)alloc((size_t)N * 100 * 4);
  float* pooled  = (float*)alloc(3200 * 4);
  float* gcnt    = (float*)alloc(32 * 4);

  auto pack = [&](const float* W, int K, int Nn) -> _Float16* {
    int KT = (K + 31) / 32, NT2 = (Nn + 15) / 16;
    size_t total = (size_t)KT * NT2 * 512;
    _Float16* P = (_Float16*)alloc(total * 2);
    int blocks = (int)((total + 255) / 256);
    k_pack<<<dim3(blocks), dim3(256), 0, stream>>>(W, P, K, Nn, KT, NT2);
    return P;
  };
  auto zero = [&](float* p, size_t n) {
    int blocks = (int)((n + 255) / 256);
    k_zero<<<dim3(blocks), dim3(256), 0, stream>>>(p, (long long)n);
  };

  // ---- parameter leaf indices (JAX pytree flatten: sorted dict keys) ----
  const float *mwh[5][3], *mwsb[5][3], *mwsvb[5][3], *mwv[5][3];
  _Float16 *mws[5][3], *mwsv[5][3];
  const float *fwh[5][2], *fwsb[5][2], *fwsvb[5][2], *fwv[5][2];
  _Float16 *fws[5][2], *fwsv[5][2];
  const float *n0g[5], *n0b[5], *n1g[5], *n1b[5];
  for (int l = 0; l < 5; ++l) {
    int b0 = 10 + 34 * l;
    fwh[l][0]  = F(b0 + 0);  fwsb[l][0]  = F(b0 + 1);
    fws[l][0]  = pack(F(b0 + 2), 132, 400);
    fwsvb[l][0] = F(b0 + 3); fwsv[l][0] = pack(F(b0 + 4), 400, 32);
    fwv[l][0]  = F(b0 + 5);
    fwh[l][1]  = F(b0 + 6);  fwsb[l][1]  = F(b0 + 7);
    fws[l][1]  = pack(F(b0 + 8), 432, 100);
    fwsvb[l][1] = F(b0 + 9); fwsv[l][1] = pack(F(b0 + 10), 100, 16);
    fwv[l][1]  = F(b0 + 11);
    for (int s = 0; s < 3; ++s) {
      int mb = b0 + 12 + 6 * s;
      mwh[l][s]  = F(mb + 0); mwsb[l][s]  = F(mb + 1);
      mws[l][s]  = pack(F(mb + 2), s == 0 ? 265 : 116, 100);
      mwsvb[l][s] = F(mb + 3); mwsv[l][s] = pack(F(mb + 4), 100, 16);
      mwv[l][s]  = F(mb + 5);
    }
    n0b[l] = F(b0 + 30); n0g[l] = F(b0 + 31);
    n1b[l] = F(b0 + 32); n1g[l] = F(b0 + 33);
  }
  _Float16* pwsOut = pack(F(190), 116, 100);

  // ---- pipeline ----
  zero(cnt, N); zero(gcnt, 32); zero(pooled, 3200);
  k_cnt<<<dim3((E + 255) / 256), dim3(256), 0, stream>>>(eidx, cnt, E);
  k_gcnt<<<dim3((N + 255) / 256), dim3(256), 0, stream>>>(batch, gcnt, N);
  k_node_embed<<<dim3((N + 127) / 128), dim3(128), 0, stream>>>(
      atoms, F(9), F(196), F(195), F(194), F(193), hs, hv, N);
  k_edge_embed<<<dim3((E + 127) / 128), dim3(128), 0, stream>>>(
      edgeS, edgeV, F(187), F(186), F(180), F(182), F(181),
      F(184), F(183), F(185), esb, evb, E);

  for (int l = 0; l < 5; ++l) {
    zero(aggS, (size_t)N * 100);
    zero(aggV, (size_t)N * 48);
    k_edge_msg<<<dim3(E / 32), dim3(128), 0, stream>>>(
        hs, hv, esb, evb, eidx, E,
        mwh[l][0], mws[l][0], mwsb[l][0], mwsv[l][0], mwsvb[l][0], mwv[l][0],
        mwh[l][1], mws[l][1], mwsb[l][1], mwsv[l][1], mwsvb[l][1], mwv[l][1],
        mwh[l][2], mws[l][2], mwsb[l][2], mwsv[l][2], mwsvb[l][2], mwv[l][2],
        aggS, aggV);
    k_ln<<<dim3((N + 127) / 128), dim3(128), 0, stream>>>(
        n0g[l], n0b[l], hs, hv, aggS, aggV, cnt, hs, hv, N);
    k_ff<<<dim3(N / 32), dim3(128), 0, stream>>>(
        hs, hv,
        fwh[l][0], fws[l][0], fwsb[l][0], fwsv[l][0], fwsvb[l][0], fwv[l][0],
        fwh[l][1], fws[l][1], fwsb[l][1], fwsv[l][1], fwsvb[l][1], fwv[l][1],
        aggS, aggV, N);
    k_ln<<<dim3((N + 127) / 128), dim3(128), 0, stream>>>(
        n1g[l], n1b[l], hs, hv, aggS, aggV, nullptr, hs, hv, N);
  }

  // output LN -> GVP -> pool -> dense
  k_ln<<<dim3((N + 127) / 128), dim3(128), 0, stream>>>(
      F(192), F(191), hs, hv, nullptr, nullptr, nullptr, aggS, aggV, N);
  k_wout<<<dim3(N / 16), dim3(128), 0, stream>>>(
      aggS, aggV, F(188), pwsOut, F(189), nodeOut, N);
  k_pool<<<dim3((N + 127) / 128), dim3(128), 0, stream>>>(nodeOut, batch, pooled, N);
  k_dense<<<dim3(1), dim3(256), 0, stream>>>(
      pooled, gcnt, F(7), F(5), F(8), F(6), (float*)d_out);
}
